// Attention_18408229830680
// MI455X (gfx1250) — matrix operand order repaired and checked by reading, not compiled, not run on hardware
//
#include <hip/hip_runtime.h>
#include <hip/hip_bf16.h>

typedef __bf16   v16bf __attribute__((ext_vector_type(16)));
typedef float    v8f   __attribute__((ext_vector_type(8)));
typedef unsigned u32;
typedef u32      v4u   __attribute__((ext_vector_type(4)));
typedef u32      v8u   __attribute__((ext_vector_type(8)));

#define B_  64
#define T_  2000
#define QD  1024
#define MD  512
#define AD  128
#define NF  32
#define KW  31

// workspace byte offsets
#define WS_PQ      0            // 64*128 f32              = 32768
#define WS_WMPACK  32768        // 8*16*32*16 bf16         = 131072
#define WS_WLPACK  163840       // 8*32*16 bf16            = 8192
#define WS_ENERGY  172032       // 64*2000 f32             = 512000
#define WS_PART    684032       // 8*64*512 f32            = 1048576

// hardware bf16 convert (backend emits v_cvt_pk_bf16_f32 / v_cvt_bf16)
static __device__ __forceinline__ __bf16 f2bf(float f) { return (__bf16)f; }

static __device__ __forceinline__ v16bf mk_afrag(float4 x0, float4 x1, float4 y0, float4 y1) {
    v16bf a;
    a[0]  = f2bf(x0.x); a[1]  = f2bf(x0.y); a[2]  = f2bf(x0.z); a[3]  = f2bf(x0.w);
    a[4]  = f2bf(x1.x); a[5]  = f2bf(x1.y); a[6]  = f2bf(x1.z); a[7]  = f2bf(x1.w);
    a[8]  = f2bf(y0.x); a[9]  = f2bf(y0.y); a[10] = f2bf(y0.z); a[11] = f2bf(y0.w);
    a[12] = f2bf(y1.x); a[13] = f2bf(y1.y); a[14] = f2bf(y1.z); a[15] = f2bf(y1.w);
    return a;
}

// ---------------------------------------------------------------------------
// Pack Wm [512,128] and Wloc [32,128] into bf16 B-fragment layout:
// frag index = ((atile*KCHUNKS + kchunk)*32 + lane)*16 + e
// B layout (32x16 bf16): lane&15 = N, lane>>4 = half, K = half*8 + (e&7) + (e>>3)*16
// ---------------------------------------------------------------------------
__global__ void k_pack(const float* __restrict__ Wm, const float* __restrict__ Wloc,
                       __bf16* __restrict__ wmp, __bf16* __restrict__ wlp) {
    int i = blockIdx.x * 256 + threadIdx.x;
    const int NWM = 8 * 16 * 32 * 16;
    if (i < NWM) {
        int e = i & 15, lane = (i >> 4) & 31, kc = (i >> 9) & 15, at = i >> 13;
        int k = kc * 32 + (lane >> 4) * 8 + (e & 7) + (e >> 3) * 16;
        int a = at * 16 + (lane & 15);
        wmp[i] = f2bf(Wm[k * AD + a]);
    } else {
        int j = i - NWM;
        if (j < 8 * 32 * 16) {
            int e = j & 15, lane = (j >> 4) & 31, at = j >> 9;
            int k = (lane >> 4) * 8 + (e & 7) + (e >> 3) * 16;
            int a = at * 16 + (lane & 15);
            wlp[j] = f2bf(Wloc[k * AD + a]);
        }
    }
}

// pq[b,a] = tanh(query @ Wq) — 8192 outputs, K=1024, Wq reads coalesced over a
__global__ void k_pq(const float* __restrict__ q, const float* __restrict__ Wq,
                     float* __restrict__ pq) {
    int i = blockIdx.x * 256 + threadIdx.x;
    int b = i >> 7, a = i & 127;
    float acc = 0.f;
    for (int k = 0; k < QD; ++k) acc += q[b * QD + k] * Wq[k * AD + a];
    pq[i] = tanhf(acc);
}

// ---------------------------------------------------------------------------
// Main fused kernel: per (b, 16-row t-tile):
//   TDM tensor_load_to_lds <- memory tile 16x512 fp32 (one DMA, wave 0 only)
//   LDS <- conv features 16x32 (location conv, SAME pad 15)
//   wave w (atile=w): pm = sum_k WMMA_bf16(Atile, WmPack)  (16 WMMAs, K=512)
//                     pl = WMMA_bf16(Gtile, WlocPack)      (1 WMMA,  K=32)
//   energy[t] = sum_a tanh(pq + tanh(pl) + tanh(pm)) * V[a]
// ---------------------------------------------------------------------------
__global__ __launch_bounds__(256) void k_main(
        const float* __restrict__ mem, const float* __restrict__ awc,
        const float* __restrict__ ck,  const float* __restrict__ pq,
        const float* __restrict__ V,
        const __bf16* __restrict__ wmp, const __bf16* __restrict__ wlp,
        float* __restrict__ energy) {
    __shared__ float ldsA[16 * MD];   // 32 KB fp32 memory tile (TDM destination)
    __shared__ float ldsG[16 * NF];   // 2 KB conv features
    __shared__ float ldsE[16];        // per-row energy accumulators

    const int tid = threadIdx.x;
    const int b   = blockIdx.y;
    const int t0  = blockIdx.x * 16;

    if (tid < 16) ldsE[tid] = 0.f;

    // ---- TDM: stage memory[b, t0:t0+16, :] into LDS with one tensor DMA ----
    // D# group0: count=1 | lds_addr | global_addr[56:0] | type=2
    // D# group1: data_size=4B, tensor_dim0=512, tensor_dim1=16,
    //            tile_dim0=512, tile_dim1=16, tensor_dim0_stride=512
    if (tid < 32) {   // one wave issues the DMA (TENSOR ops ignore EXEC)
        unsigned long long gaddr =
            (unsigned long long)(const void*)(mem + (size_t)(b * T_ + t0) * MD);
        v4u g0;
        g0.x = 1u;                                   // count=1, user mode
        g0.y = (u32)(size_t)(void*)ldsA;             // LDS byte address ([31:0] of flat)
        g0.z = (u32)gaddr;                           // global addr [31:0]
        g0.w = (u32)((gaddr >> 32) & 0x01FFFFFFull)  // global addr [56:32]
             | (2u << 30);                           // type=2 ("image")
        v8u g1;
        g1[0] = 2u << 16;        // workgroup_mask=0, data_size=2 (4 bytes)
        g1[1] = 512u << 16;      // tensor_dim0[15:0] in [31:16]
        g1[2] = 16u << 16;       // tensor_dim0 hi=0, tensor_dim1[15:0] in [31:16]
        g1[3] = 512u << 16;      // tensor_dim1 hi=0, tile_dim0=512 in [31:16]
        g1[4] = 16u;             // tile_dim1=16, tile_dim2=0
        g1[5] = 512u;            // tensor_dim0_stride[31:0] = 512
        g1[6] = 0u;              // stride hi, tensor_dim1_stride lo
        g1[7] = 0u;              // tensor_dim1_stride hi
        asm volatile("tensor_load_to_lds %0, %1" :: "s"(g0), "s"(g1) : "memory");
        __builtin_amdgcn_s_wait_tensorcnt(0);
    }

    // location conv: G[m,f] = sum_{k,c} awc[b,c,t0+m+k-15] * ck[k,c,f]
    for (int i = tid; i < 16 * NF; i += 256) {
        int m = i >> 5, f = i & 31;
        int t = t0 + m;
        float acc = 0.f;
        for (int k = 0; k < KW; ++k) {
            int tt = t + k - 15;
            if (tt >= 0 && tt < T_) {
                acc += awc[(b * 2 + 0) * T_ + tt] * ck[(k * 2 + 0) * NF + f];
                acc += awc[(b * 2 + 1) * T_ + tt] * ck[(k * 2 + 1) * NF + f];
            }
        }
        ldsG[i] = acc;
    }
    __syncthreads();

    const int lane = tid & 31;
    const int wid  = tid >> 5;              // atile 0..7
    const int m    = lane & 15;             // A row (t within tile)
    const int kb   = (lane < 16) ? 0 : 8;   // A K-base per ISA 16-bit A layout

    // ---- pm = memory_tile @ Wm (K = 512 = 16 chunks of 32) ----
    v8f accM = {};
    const v16bf* bM = (const v16bf*)wmp + (size_t)(wid * 16) * 32;
    #pragma unroll
    for (int kc = 0; kc < 16; ++kc) {
        const float4* pa = (const float4*)&ldsA[m * MD + kc * 32 + kb];
        float4 x0 = pa[0], x1 = pa[1];      // K = kb .. kb+7
        float4 y0 = pa[4], y1 = pa[5];      // K = kb+16 .. kb+23
        v16bf af = mk_afrag(x0, x1, y0, y1);
        v16bf bf = bM[kc * 32 + lane];
        if (kc + 1 < 16) __builtin_prefetch(&bM[(kc + 1) * 32 + lane], 0, 0);
        accM = __builtin_amdgcn_wmma_f32_16x16x32_bf16(
                   false, af, false, bf, (short)0, accM, false, false);
    }

    // ---- ploc = conv_tile @ Wloc (K = 32 = exactly one WMMA) ----
    v8f accL = {};
    {
        const float4* pg = (const float4*)&ldsG[m * NF + kb];
        float4 x0 = pg[0], x1 = pg[1];
        float4 y0 = pg[4], y1 = pg[5];
        v16bf af = mk_afrag(x0, x1, y0, y1);
        v16bf bf = ((const v16bf*)wlp)[wid * 32 + lane];
        accL = __builtin_amdgcn_wmma_f32_16x16x32_bf16(
                   false, af, false, bf, (short)0, accL, false, false);
    }

    // ---- energies: C/D frag layout: VGPR r -> row (r + 8*(lane>=16)), col = lane&15
    const int a   = wid * 16 + (lane & 15);
    const float pqv = pq[b * AD + a];
    const float Vv  = V[a];
    #pragma unroll
    for (int r = 0; r < 8; ++r) {
        float s = tanhf(pqv + tanhf(accL[r]) + tanhf(accM[r])) * Vv;
        s += __shfl_xor(s, 8, 16);
        s += __shfl_xor(s, 4, 16);
        s += __shfl_xor(s, 2, 16);
        s += __shfl_xor(s, 1, 16);
        if ((lane & 15) == 0)
            atomicAdd(&ldsE[r + (lane >> 4) * 8], s);
    }
    __syncthreads();
    if (tid < 16) energy[b * T_ + t0 + tid] = ldsE[tid];
}

// softmax over T per batch row (one workgroup per b)
__global__ __launch_bounds__(256) void k_softmax(const float* __restrict__ energy,
                                                 float* __restrict__ wout) {
    __shared__ float red[256];
    const int b = blockIdx.x, tid = threadIdx.x;
    float mx = -1e30f;
    for (int t = tid; t < T_; t += 256) mx = fmaxf(mx, energy[b * T_ + t]);
    red[tid] = mx; __syncthreads();
    for (int s = 128; s > 0; s >>= 1) {
        if (tid < s) red[tid] = fmaxf(red[tid], red[tid + s]);
        __syncthreads();
    }
    mx = red[0]; __syncthreads();
    float sum = 0.f;
    for (int t = tid; t < T_; t += 256) sum += __expf(energy[b * T_ + t] - mx);
    red[tid] = sum; __syncthreads();
    for (int s = 128; s > 0; s >>= 1) {
        if (tid < s) red[tid] += red[tid + s];
        __syncthreads();
    }
    const float inv = 1.0f / red[0];
    for (int t = tid; t < T_; t += 256)
        wout[b * T_ + t] = __expf(energy[b * T_ + t] - mx) * inv;
}

// context partials: grid (b, tchunk of 250); coalesced over d
__global__ __launch_bounds__(256) void k_ctx_part(const float* __restrict__ mem,
                                                  const float* __restrict__ w,
                                                  float* __restrict__ part) {
    const int b = blockIdx.x, c = blockIdx.y, tid = threadIdx.x;
    const int tstart = c * 250, tend = tstart + 250;
    const int d0 = tid, d1 = tid + 256;
    float acc0 = 0.f, acc1 = 0.f;
    for (int t = tstart; t < tend; ++t) {
        const float wv = w[b * T_ + t];
        const float* row = mem + (size_t)(b * T_ + t) * MD;
        acc0 += wv * row[d0];
        acc1 += wv * row[d1];
    }
    part[((size_t)c * B_ + b) * MD + d0] = acc0;
    part[((size_t)c * B_ + b) * MD + d1] = acc1;
}

__global__ void k_ctx_red(const float* __restrict__ part, float* __restrict__ ctx) {
    int i = blockIdx.x * 256 + threadIdx.x;  // i = b*512 + d, 32768 total
    float acc = 0.f;
    #pragma unroll
    for (int c = 0; c < 8; ++c) acc += part[(size_t)c * B_ * MD + i];
    ctx[i] = acc;
}

extern "C" void kernel_launch(void* const* d_in, const int* in_sizes, int n_in,
                              void* d_out, int out_size, void* d_ws, size_t ws_size,
                              hipStream_t stream) {
    (void)in_sizes; (void)n_in; (void)out_size; (void)ws_size;
    const float* query  = (const float*)d_in[0];
    const float* memory = (const float*)d_in[1];
    const float* awc    = (const float*)d_in[2];
    const float* Wq     = (const float*)d_in[3];
    const float* Wm     = (const float*)d_in[4];
    const float* ck     = (const float*)d_in[5];
    const float* Wloc   = (const float*)d_in[6];
    const float* V      = (const float*)d_in[7];

    char* ws = (char*)d_ws;
    float*  pq     = (float*)(ws + WS_PQ);
    __bf16* wmp    = (__bf16*)(ws + WS_WMPACK);
    __bf16* wlp    = (__bf16*)(ws + WS_WLPACK);
    float*  energy = (float*)(ws + WS_ENERGY);
    float*  part   = (float*)(ws + WS_PART);

    float* ctx  = (float*)d_out;             // [64,512]
    float* wout = (float*)d_out + B_ * MD;   // [64,2000]

    const int npack = 8 * 16 * 32 * 16 + 8 * 32 * 16;
    k_pack<<<(npack + 255) / 256, 256, 0, stream>>>(Wm, Wloc, wmp, wlp);
    k_pq<<<(B_ * AD) / 256, 256, 0, stream>>>(query, Wq, pq);
    k_main<<<dim3(T_ / 16, B_), 256, 0, stream>>>(memory, awc, ck, pq, V, wmp, wlp, energy);
    k_softmax<<<B_, 256, 0, stream>>>(energy, wout);
    k_ctx_part<<<dim3(B_, 8), 256, 0, stream>>>(memory, wout, part);
    k_ctx_red<<<(B_ * MD) / 256, 256, 0, stream>>>(part, ctx);
}
